// GraphResnetBottleneckBlock_37855841747676
// MI455X (gfx1250) — matrix operand order repaired and checked by reading, not compiled, not run on hardware
//
#include <hip/hip_runtime.h>
#include <hip/hip_bf16.h>

// ---------------------------------------------------------------------------
// KPConv-style bottleneck on MI455X (gfx1250, wave32, WMMA + TDM).
// Heavy math: v_wmma_f32_16x16x32_f16 (f16 in, fp32 accum).
// Conv weight panel is DMA'd into LDS once per block via tensor_load_to_lds
// (TENSORcnt / s_wait_tensorcnt), then kept resident across the K loop.
// BN batch statistics use deterministic fixed-order partial sums (no atomics).
// ---------------------------------------------------------------------------

typedef _Float16 half_t;
typedef __attribute__((ext_vector_type(16))) _Float16 v16h;
typedef __attribute__((ext_vector_type(8)))  _Float16 v8h;
typedef __attribute__((ext_vector_type(8)))  float    v8f;
typedef __attribute__((ext_vector_type(4)))  unsigned int v4u;
typedef __attribute__((ext_vector_type(8)))  int      v8i;
typedef __attribute__((ext_vector_type(4)))  int      v4i;

#define NPTS   30000
#define KNBR   32
#define INDIM  256
#define HID    128
#define OUTDIM 512
#define KDPAD  160      // HID+3 = 131 padded to 5*32 for the WMMA K loop
#define LEAK   0.1f
#define EPS_BN 1e-5f

// ---------------- WMMA fragment helpers (ISA 7.12.2 layouts, wave32) -------

static __device__ __forceinline__ v16h pack16(v8h a, v8h b) {
  v16h r;
#pragma unroll
  for (int i = 0; i < 8; ++i) { r[i] = a[i]; r[i + 8] = b[i]; }
  return r;
}

// A fragment: 16x32 f16, lane L holds row M=L%16; K halves split
// {0..7 | 8..15} by lane half in v0..3, +16 in v4..7.  LDS tile [M][32].
static __device__ __forceinline__ v16h ldsA_frag(const half_t* sA, int mBase, int lane) {
  const int lr = lane & 15, hi = (lane >> 4) & 1;
  const half_t* row = sA + (mBase + lr) * 32;
  v8h lo = *(const v8h*)(row + hi * 8);
  v8h hh = *(const v8h*)(row + 16 + hi * 8);
  return pack16(lo, hh);
}

// B fragment: 32x16 f16, lane L holds column N=L%16; K halves split
// {0..7 | 8..15} by lane half in v0..3, +16 in v4..7.  LDS tile stored TRANSPOSED with row stride ldk.
static __device__ __forceinline__ v16h ldsB_frag(const half_t* sBt, int nBase,
                                                 int k0, int lane, int ldk) {
  const int ln = lane & 15, hi = (lane >> 4) & 1;
  const half_t* row = sBt + (nBase + ln) * ldk + k0;
  v8h b0 = *(const v8h*)(row + hi * 8);
  v8h b1 = *(const v8h*)(row + 16 + hi * 8);
  return pack16(b0, b1);
}

static __device__ __forceinline__ v8f wmma32(v16h a, v16h b, v8f c) {
  return __builtin_amdgcn_wmma_f32_16x16x32_f16(false, a, false, b,
                                                (short)0, c, false, false);
}

// ---------------- generic tiled f16 GEMM: C[M,Nc] = A[M,Kd] * B[Kd,Nc] -----
// block tile 32x64, 8 waves (2x4), one 16x16 C tile per wave, K step 32.
__global__ void k_gemm_f16(const half_t* __restrict__ A, const half_t* __restrict__ B,
                           float* __restrict__ C, int M, int Kd, int Nc) {
  __shared__ half_t sA[32 * 32];
  __shared__ half_t sBt[64 * 32];
  const int tid = threadIdx.x, lane = tid & 31, wave = tid >> 5;
  const int waveM = wave >> 2, waveN = wave & 3;
  const int gm0 = blockIdx.y * 32, gn0 = blockIdx.x * 64;
  v8f acc = {};

  for (int k0 = 0; k0 < Kd; k0 += 32) {
    { // stage A tile 32x32; clamp OOB rows (results discarded at store)
      int r = tid >> 3, c4 = (tid & 7) * 4;
      int gr = min(gm0 + r, M - 1);
      *(uint2*)&sA[r * 32 + c4] = *(const uint2*)(A + (size_t)gr * Kd + k0 + c4);
    }
    { // stage B tile transposed: sBt[n][k]
      int n = tid & 63, kh = (tid >> 6) * 8;
#pragma unroll
      for (int j = 0; j < 8; ++j)
        sBt[n * 32 + kh + j] = B[(size_t)(k0 + kh + j) * Nc + gn0 + n];
    }
    __syncthreads();
    v16h af = ldsA_frag(sA, waveM * 16, lane);
    v16h bf = ldsB_frag(sBt, waveN * 16, 0, lane, 32);
    acc = wmma32(af, bf, acc);
    __syncthreads();
  }

  const int col = gn0 + waveN * 16 + (lane & 15);
  const int hi = (lane >> 4) & 1;
  const int rbase = gm0 + waveM * 16 + hi * 8;
  float* Cp = C + (size_t)rbase * Nc + col;
  if (gm0 + 32 <= M) {            // block-uniform fast path: no per-row guards
#pragma unroll
    for (int i = 0; i < 8; ++i) Cp[(size_t)i * Nc] = acc[i];
  } else {
#pragma unroll
    for (int i = 0; i < 8; ++i) if (rbase + i < M) Cp[(size_t)i * Nc] = acc[i];
  }
}

// ---------------- neighbor-gather conv GEMM (stats pass / apply pass) ------
// Block = 2 points = 64 rows of the flattened [N*K, 131] matrix, Nc = 128.
// 8 waves arranged 4(M) x 2(N); each wave: one 16-row M tile x 4 N tiles.
// Weight panel loaded once per block by the Tensor Data Mover, kept resident.
template <bool STATS>
__global__ void k_conv(const half_t* __restrict__ xg,     // (N+1) x KDPAD f16, padded
                       const float*  __restrict__ points, // N x 3
                       const int*    __restrict__ nbr,    // N x K
                       const half_t* __restrict__ Wp,     // KDPAD x HID f16 (zero padded)
                       const float*  __restrict__ scale2, // HID (apply pass)
                       const float*  __restrict__ bias2,  // HID (apply pass)
                       float* __restrict__ psum, float* __restrict__ psq, // stats pass
                       half_t* __restrict__ ymaxH) {      // N x HID (apply pass)
  __shared__ half_t sA[64 * 32];                          // gathered A K-step tile
  __shared__ half_t sBt[HID * KDPAD];                     // resident transposed weights
  __shared__ union { half_t wrow[KDPAD * HID]; float ysh[64 * HID]; } u;

  const int tid = threadIdx.x, lane = tid & 31, wave = tid >> 5;
  const int waveM = wave >> 1, waveN = wave & 1;
  const int p0 = blockIdx.x * 2;

  // ---- TDM: DMA the 40KB padded weight panel (row-major) into LDS --------
  if (tid == 0) {
    unsigned lds_addr = (unsigned)(unsigned long long)(&u.wrow[0]);
    unsigned long long ga = (unsigned long long)(const void*)Wp;
    const unsigned NEL = KDPAD * HID;                     // 20480 f16 elements
    v4u g0;
    g0[0] = 1u;                                           // count=1, user mode
    g0[1] = lds_addr;                                     // lds_addr [63:32]
    g0[2] = (unsigned)(ga & 0xffffffffu);                 // global_addr lo
    g0[3] = (unsigned)((ga >> 32) & 0x01ffffffu) | (2u << 30); // addr hi | type=2
    v8i g1;
    g1[0] = (int)(1u << 16);                              // data_size = 2 bytes
    g1[1] = (int)((NEL & 0xffffu) << 16);                 // tensor_dim0 lo16
    g1[2] = (int)((NEL >> 16) | (1u << 16));              // tensor_dim0 hi | tensor_dim1=1
    g1[3] = (int)((NEL & 0xffffu) << 16);                 // tile_dim0
    g1[4] = 1;                                            // tile_dim1=1, tile_dim2=0
    g1[5] = (int)NEL;                                     // tensor_dim0_stride lo
    g1[6] = 0;
    g1[7] = 0;
    v4i z4 = {0, 0, 0, 0};
#if defined(__clang_major__) && __clang_major__ >= 23
    v8i z8 = {0, 0, 0, 0, 0, 0, 0, 0};
    __builtin_amdgcn_tensor_load_to_lds(g0, g1, z4, z4, z8, 0);
#else
    __builtin_amdgcn_tensor_load_to_lds(g0, g1, z4, z4, 0);
#endif
    __builtin_amdgcn_s_wait_tensorcnt(0);                 // s_wait_tensorcnt 0
  }
  __syncthreads();

  // transpose weights in LDS: sBt[n*KDPAD + k] = wrow[k*HID + n]
  for (int idx = tid; idx < KDPAD * HID; idx += 256) {
    int k = idx >> 7, n = idx & 127;
    sBt[n * KDPAD + k] = u.wrow[idx];
  }

  // per-thread gather assignment: 4 threads per A row
  const int ar  = tid >> 2;            // 0..63 : row in tile (= point*32 + k)
  const int ac0 = (tid & 3) * 8;       // 8 halves per thread
  const int ap  = p0 + (ar >> 5);
  const int ak  = ar & 31;
  const int ind = nbr[ap * KNBR + ak]; // in [0, NPTS]; NPTS = shadow row
  float pdx[3];
#pragma unroll
  for (int d = 0; d < 3; ++d)
    pdx[d] = (ind < NPTS) ? (points[(size_t)ind * 3 + d] - points[(size_t)ap * 3 + d]) : 0.f;

  v8f zero = {};
  v8f acc[4] = {zero, zero, zero, zero};
  __syncthreads();                      // weights resident

  for (int k0 = 0; k0 < KDPAD; k0 += 32) {
    // stage gathered A tile: one aligned 16B vector load per thread
    v8h av = *(const v8h*)(xg + (size_t)ind * KDPAD + k0 + ac0);
    *(v8h*)&sA[ar * 32 + ac0] = av;
    if (k0 == 0 && ac0 == 0) {          // patch geometry (cols 0..2) once
      sA[ar * 32 + 0] = (half_t)pdx[0];
      sA[ar * 32 + 1] = (half_t)pdx[1];
      sA[ar * 32 + 2] = (half_t)pdx[2];
    }
    __syncthreads();
    v16h af = ldsA_frag(sA, waveM * 16, lane);
#pragma unroll
    for (int nt = 0; nt < 4; ++nt) {
      v16h bf = ldsB_frag(sBt, waveN * 64 + nt * 16, k0, lane, KDPAD);
      acc[nt] = wmma32(af, bf, acc[nt]);
    }
    __syncthreads();
  }

  // spill C tiles to LDS for the cross-row epilogues (overlays wrow, done)
  const int hi = (lane >> 4) & 1;
#pragma unroll
  for (int nt = 0; nt < 4; ++nt) {
    int col = waveN * 64 + nt * 16 + (lane & 15);
#pragma unroll
    for (int i = 0; i < 8; ++i) {
      int row = waveM * 16 + (hi ? 8 + i : i);
      u.ysh[row * 128 + col] = acc[nt][i];
    }
  }
  __syncthreads();

  if (STATS) {
    // per-channel sum / sumsq over the 64 (n,k) rows -> fixed-order partials
    if (tid < HID) {
      float s = 0.f, q = 0.f;
#pragma unroll 4
      for (int r = 0; r < 64; ++r) { float y = u.ysh[r * 128 + tid]; s += y; q += y * y; }
      psum[(size_t)blockIdx.x * HID + tid] = s;
      psq [(size_t)blockIdx.x * HID + tid] = q;
    }
  } else {
    // BN2 + leaky + max over K, per (point, channel)
    int pp = tid >> 7, c = tid & 127;
    float sc = scale2[c], bi = bias2[c];
    float m = -3.0e38f;
    for (int r = 0; r < KNBR; ++r) {
      float y = u.ysh[(pp * 32 + r) * 128 + c] * sc + bi;
      y = (y >= 0.f) ? y : LEAK * y;
      m = fmaxf(m, y);
    }
    ymaxH[(size_t)(p0 + pp) * HID + c] = (half_t)m;
  }
}

// ---------------- deterministic column statistics --------------------------
__global__ void k_colstats(const float* __restrict__ X, int M, int C,
                           float* __restrict__ psum, float* __restrict__ psq) {
  const int b = blockIdx.x;
  const int r0 = b * 256;
  const int r1 = min(M, r0 + 256);
  for (int c = threadIdx.x; c < C; c += blockDim.x) {
    float s = 0.f, q = 0.f;
    for (int r = r0; r < r1; ++r) { float v = X[(size_t)r * C + c]; s += v; q += v * v; }
    psum[(size_t)b * C + c] = s;
    psq [(size_t)b * C + c] = q;
  }
}

__global__ void k_bn_finalize(const float* __restrict__ psum, const float* __restrict__ psq,
                              int nb, int C, float cnt,
                              const float* __restrict__ g, const float* __restrict__ b,
                              float* __restrict__ scale, float* __restrict__ bias) {
  int c = blockIdx.x * blockDim.x + threadIdx.x;
  if (c >= C) return;
  float s = 0.f, q = 0.f;
  for (int i = 0; i < nb; ++i) { s += psum[(size_t)i * C + c]; q += psq[(size_t)i * C + c]; }
  float mean = s / cnt;
  float var  = q / cnt - mean * mean;          // biased variance (matches jnp.var)
  float sc   = g[c] * rsqrtf(var + EPS_BN);
  scale[c] = sc;
  bias[c]  = b[c] - mean * sc;
}

// ---------------- small elementwise kernels --------------------------------
__global__ void k_f32_to_f16(const float* __restrict__ in, half_t* __restrict__ out, int n) {
  int i = blockIdx.x * blockDim.x + threadIdx.x;
  if (i < n) out[i] = (half_t)in[i];
}

__global__ void k_pad_convw(const float* __restrict__ w, half_t* __restrict__ out) {
  int i = blockIdx.x * blockDim.x + threadIdx.x;      // KDPAD*HID
  if (i >= KDPAD * HID) return;
  int r = i / HID;
  out[i] = (r < 131) ? (half_t)w[i] : (half_t)0.f;
}

// BN1 apply + leaky; output padded to KDPAD-wide rows: cols 0..2 zero (geometry
// slots), 3..130 activated features, 131..159 zero; shadow row N all-zero.
__global__ void k_bn1_apply(const float* __restrict__ t1, const float* __restrict__ sc,
                            const float* __restrict__ bi, half_t* __restrict__ xg) {
  int i = blockIdx.x * blockDim.x + threadIdx.x;      // (N+1)*KDPAD
  if (i >= (NPTS + 1) * KDPAD) return;
  int n = i / KDPAD, c = i - n * KDPAD;
  float v = 0.f;
  if (n < NPTS && c >= 3 && c < 131) {
    int cc = c - 3;
    v = t1[(size_t)n * HID + cc] * sc[cc] + bi[cc];
    v = (v >= 0.f) ? v : LEAK * v;
  }
  xg[i] = (half_t)v;
}

__global__ void k_final(const float* __restrict__ t3, const float* __restrict__ tsc,
                        const float* __restrict__ s3, const float* __restrict__ b3,
                        const float* __restrict__ ss, const float* __restrict__ bs,
                        float* __restrict__ out) {
  int i = blockIdx.x * blockDim.x + threadIdx.x;      // N*OUTDIM
  if (i >= NPTS * OUTDIM) return;
  int c = i & (OUTDIM - 1);
  float v = t3[i] * s3[c] + b3[c] + tsc[i] * ss[c] + bs[c];
  out[i] = (v >= 0.f) ? v : LEAK * v;
}

// ---------------- host orchestration ---------------------------------------
extern "C" void kernel_launch(void* const* d_in, const int* in_sizes, int n_in,
                              void* d_out, int out_size, void* d_ws, size_t ws_size,
                              hipStream_t stream) {
  (void)in_sizes; (void)n_in; (void)out_size; (void)ws_size;

  const float* features = (const float*)d_in[0];   // [N,256]
  const float* points   = (const float*)d_in[1];   // [N,3]
  const int*   nbr      = (const int*)  d_in[2];   // [N,32]
  const float* w1       = (const float*)d_in[3];   // [256,128]
  const float* g1       = (const float*)d_in[4];
  const float* b1       = (const float*)d_in[5];
  const float* conv_w   = (const float*)d_in[6];   // [131,128]
  const float* g2       = (const float*)d_in[7];
  const float* b2       = (const float*)d_in[8];
  const float* w2       = (const float*)d_in[9];   // [128,512]
  const float* g3       = (const float*)d_in[10];
  const float* b3       = (const float*)d_in[11];
  const float* ws       = (const float*)d_in[12];  // [256,512]
  const float* gs       = (const float*)d_in[13];
  const float* bs       = (const float*)d_in[14];
  float* out = (float*)d_out;

  // bump allocator on d_ws (total ~188 MB)
  char* wp = (char*)d_ws;
  auto alloc = [&](size_t bytes) -> char* {
    char* p = wp; wp += (bytes + 255) & ~(size_t)255; return p;
  };
  half_t* featH  = (half_t*)alloc((size_t)NPTS * INDIM * 2);          // 15.4 MB
  half_t* w1H    = (half_t*)alloc((size_t)INDIM * HID * 2);
  half_t* convWH = (half_t*)alloc((size_t)KDPAD * HID * 2);
  half_t* w2H    = (half_t*)alloc((size_t)HID * OUTDIM * 2);
  half_t* wsH    = (half_t*)alloc((size_t)INDIM * OUTDIM * 2);
  half_t* xg     = (half_t*)alloc((size_t)(NPTS + 1) * KDPAD * 2);    // 9.6 MB
  half_t* ymaxH  = (half_t*)alloc((size_t)NPTS * HID * 2);            // 7.7 MB
  float*  t1     = (float*)alloc((size_t)NPTS * HID * 4);             // 15.4 MB
  float*  t3     = (float*)alloc((size_t)NPTS * OUTDIM * 4);          // 61.4 MB
  float*  tsc    = (float*)alloc((size_t)NPTS * OUTDIM * 4);          // 61.4 MB
  const int convBlocks = NPTS / 2;                                    // 15000
  float*  psum   = (float*)alloc((size_t)convBlocks * HID * 4);       // 7.7 MB (reused)
  float*  psq    = (float*)alloc((size_t)convBlocks * HID * 4);       // 7.7 MB (reused)
  float*  sc1 = (float*)alloc(HID * 4);     float* bi1 = (float*)alloc(HID * 4);
  float*  sc2 = (float*)alloc(HID * 4);     float* bi2 = (float*)alloc(HID * 4);
  float*  sc3 = (float*)alloc(OUTDIM * 4);  float* bi3 = (float*)alloc(OUTDIM * 4);
  float*  scS = (float*)alloc(OUTDIM * 4);  float* biS = (float*)alloc(OUTDIM * 4);

  const int T = 256;
  const int mBlocks = (NPTS + 31) / 32;            // 938
  const int sBlocks = (NPTS + 255) / 256;          // 118 row-blocks for col stats

  // 0) f32 -> f16 conversions
  k_f32_to_f16<<<(NPTS * INDIM + T - 1) / T, T, 0, stream>>>(features, featH, NPTS * INDIM);
  k_f32_to_f16<<<(INDIM * HID + T - 1) / T, T, 0, stream>>>(w1, w1H, INDIM * HID);
  k_pad_convw <<<(KDPAD * HID + T - 1) / T, T, 0, stream>>>(conv_w, convWH);
  k_f32_to_f16<<<(HID * OUTDIM + T - 1) / T, T, 0, stream>>>(w2, w2H, HID * OUTDIM);
  k_f32_to_f16<<<(INDIM * OUTDIM + T - 1) / T, T, 0, stream>>>(ws, wsH, INDIM * OUTDIM);

  // 1) unary1 GEMM + BN1 stats + apply (writes padded f16 x with shadow row)
  k_gemm_f16<<<dim3(HID / 64, mBlocks), T, 0, stream>>>(featH, w1H, t1, NPTS, INDIM, HID);
  k_colstats<<<sBlocks, T, 0, stream>>>(t1, NPTS, HID, psum, psq);
  k_bn_finalize<<<1, 128, 0, stream>>>(psum, psq, sBlocks, HID, (float)NPTS, g1, b1, sc1, bi1);
  k_bn1_apply<<<((NPTS + 1) * KDPAD + T - 1) / T, T, 0, stream>>>(t1, sc1, bi1, xg);

  // 2) neighbor conv: stats pass -> BN2 params -> apply pass (leaky + max_K)
  k_conv<true ><<<convBlocks, T, 0, stream>>>(xg, points, nbr, convWH,
                                              nullptr, nullptr, psum, psq, nullptr);
  k_bn_finalize<<<1, 128, 0, stream>>>(psum, psq, convBlocks, HID,
                                       (float)NPTS * (float)KNBR, g2, b2, sc2, bi2);
  k_conv<false><<<convBlocks, T, 0, stream>>>(xg, points, nbr, convWH,
                                              sc2, bi2, nullptr, nullptr, ymaxH);

  // 3) unary2 + shortcut GEMMs
  k_gemm_f16<<<dim3(OUTDIM / 64, mBlocks), T, 0, stream>>>(ymaxH, w2H, t3, NPTS, HID, OUTDIM);
  k_gemm_f16<<<dim3(OUTDIM / 64, mBlocks), T, 0, stream>>>(featH, wsH, tsc, NPTS, INDIM, OUTDIM);

  // 4) BN3 / BNs stats + fused add + leaky
  k_colstats<<<sBlocks, T, 0, stream>>>(t3, NPTS, OUTDIM, psum, psq);
  k_bn_finalize<<<(OUTDIM + 127) / 128, 128, 0, stream>>>(psum, psq, sBlocks, OUTDIM,
                                                          (float)NPTS, g3, b3, sc3, bi3);
  k_colstats<<<sBlocks, T, 0, stream>>>(tsc, NPTS, OUTDIM, psum, psq);
  k_bn_finalize<<<(OUTDIM + 127) / 128, 128, 0, stream>>>(psum, psq, sBlocks, OUTDIM,
                                                          (float)NPTS, gs, bs, scS, biS);
  k_final<<<(NPTS * OUTDIM + T - 1) / T, T, 0, stream>>>(t3, tsc, sc3, bi3, scS, biS, out);
}
